// GaussianMoELayer_33724083208581
// MI455X (gfx1250) — matrix operand order repaired and checked by reading, not compiled, not run on hardware
//
#include <hip/hip_runtime.h>
#include <math.h>

// ---------------------------------------------------------------------------
// Gaussian MoE (dense-all-experts) for MI455X / gfx1250, wave32.
// Compute-bound (~1.1 TFLOP): both GEMMs run on v_wmma_f32_16x16x32_bf16.
// Per expert: GEMM1 (x[8192,1024] x W1 -> gelu -> h bf16 in ws), then
// GEMM2 (h x W2, scaled by softmax weight, accumulated into d_out).
// GEMM2's A tile (bf16 h) is staged with CDNA5 async loads to LDS
// (global_load_async_to_lds_b128, ASYNCcnt).
// ---------------------------------------------------------------------------

#define NEXP   8
#define DIN    1024
#define DH     4096
#define DOUT   1024
#define NTOK   8192

#define BM 128
#define BN 128
#define BK 64
#define MT 4   // 16x16 tiles per wave (M)
#define NT 2   // 16x16 tiles per wave (N)

typedef __bf16 v16bf __attribute__((ext_vector_type(16)));
typedef __bf16 v4bf  __attribute__((ext_vector_type(4)));
typedef float  v8f   __attribute__((ext_vector_type(8)));

union FragU { v16bf v; uint4 q[2]; };
union AccU  { v8f v; float f[8]; };

__device__ __forceinline__ void cvt_store4(__bf16* p, float4 v) {
  v4bf b;
  b[0] = (__bf16)v.x; b[1] = (__bf16)v.y; b[2] = (__bf16)v.z; b[3] = (__bf16)v.w;
  *(v4bf*)p = b;   // 8-byte ds_store
}

// LDS byte offset of a __shared__ object: generic LDS pointers carry the
// LDS offset in addr[31:0] (aperture id lives in addr[63:32]).
__device__ __forceinline__ uint32_t lds_off(const void* p) {
  return (uint32_t)(__SIZE_TYPE__)p;
}

// CDNA5 async copy: 16B global -> LDS per lane, tracked by ASYNCcnt.
__device__ __forceinline__ void async_copy_b128(uint32_t ldsByteOff, const void* gptr) {
  asm volatile("global_load_async_to_lds_b128 %0, %1, off"
               :: "v"(ldsByteOff), "v"(gptr) : "memory");
}

__device__ __forceinline__ void wait_asynccnt0() {
  asm volatile("s_wait_asynccnt 0x0" ::: "memory");
}

// ---------------------------------------------------------------------------
// GEMM1: h[row, col] = gelu( X[row,:] * W1e[:, col] + b1e[col] ), bf16 out.
// M=8192 N=4096 K=1024.  (f32 inputs: VGPR-staged convert to bf16 in LDS.)
// ---------------------------------------------------------------------------
__global__ __launch_bounds__(256) void moe_gemm1(
    const float* __restrict__ X, const float* __restrict__ W1e,
    const float* __restrict__ b1e, __bf16* __restrict__ H)
{
  __shared__ __align__(16) __bf16 Asm[BM][BK];
  __shared__ __align__(16) __bf16 Bsm[BK][BN];
  __shared__ float b1s[BN];

  const int tid    = threadIdx.x;
  const int lane   = tid & 31;
  const int lane16 = lane & 15;
  const int laneHi = lane >> 4;
  const int wid    = tid >> 5;

  const int col0 = blockIdx.x * BN;
  const int row0 = blockIdx.y * BM;
  const int wm0  = (wid >> 2) * (MT * 16);  // 0,64
  const int wn0  = (wid & 3)  * (NT * 16);  // 0,32,64,96

  if (tid < BN) b1s[tid] = b1e[col0 + tid];

  AccU acc[MT][NT];
#pragma unroll
  for (int mi = 0; mi < MT; ++mi)
#pragma unroll
    for (int ni = 0; ni < NT; ++ni)
#pragma unroll
      for (int r = 0; r < 8; ++r) acc[mi][ni].f[r] = 0.0f;

  const int ar = tid >> 4;          // 0..15
  const int ac = (tid & 15) << 2;   // 0..60
  const int br = tid >> 5;          // 0..7
  const int bc = (tid & 31) << 2;   // 0..124

  for (int k0 = 0; k0 < DIN; k0 += BK) {
    if (k0 + BK < DIN) {  // warm L2 for next K tile (global_prefetch_b8)
      __builtin_prefetch(X + (size_t)(row0 + ar) * DIN + (k0 + BK) + ac, 0, 0);
      __builtin_prefetch(W1e + (size_t)(k0 + BK + br) * DH + col0 + bc, 0, 0);
    }
#pragma unroll
    for (int rr = 0; rr < BM; rr += 16) {
      float4 v = *(const float4*)(X + (size_t)(row0 + ar + rr) * DIN + k0 + ac);
      cvt_store4(&Asm[ar + rr][ac], v);
    }
#pragma unroll
    for (int rr = 0; rr < BK; rr += 8) {
      float4 v = *(const float4*)(W1e + (size_t)(k0 + br + rr) * DH + col0 + bc);
      cvt_store4(&Bsm[br + rr][bc], v);
    }
    __syncthreads();

#pragma unroll
    for (int ks = 0; ks < BK; ks += 32) {
      FragU a[MT], b[NT];
#pragma unroll
      for (int mi = 0; mi < MT; ++mi) {
        const __bf16* p = &Asm[wm0 + mi * 16 + lane16][ks + laneHi * 8];
        a[mi].q[0] = *(const uint4*)p;
        a[mi].q[1] = *(const uint4*)(p + 16);
      }
      const int kB = ks + laneHi * 8;
#pragma unroll
      for (int ni = 0; ni < NT; ++ni) {
#pragma unroll
        for (int s = 0; s < 16; ++s)
          b[ni].v[s] = Bsm[kB + s + (s & 8)][wn0 + ni * 16 + lane16];
      }
#pragma unroll
      for (int mi = 0; mi < MT; ++mi)
#pragma unroll
        for (int ni = 0; ni < NT; ++ni)
          acc[mi][ni].v = __builtin_amdgcn_wmma_f32_16x16x32_bf16(
              false, a[mi].v, false, b[ni].v, (short)0, acc[mi][ni].v, false, false);
    }
    __syncthreads();
  }

#pragma unroll
  for (int mi = 0; mi < MT; ++mi) {
#pragma unroll
    for (int ni = 0; ni < NT; ++ni) {
      const int nl = wn0 + ni * 16 + lane16;
      const int gn = col0 + nl;
#pragma unroll
      for (int r = 0; r < 8; ++r) {
        const int gm = row0 + wm0 + mi * 16 + laneHi * 8 + r;
        float v = acc[mi][ni].f[r] + b1s[nl];
        v = 0.5f * v * (1.0f + erff(v * 0.70710678118654752f));   // exact gelu
        H[(size_t)gm * DH + gn] = (__bf16)v;
      }
    }
  }
}

// ---------------------------------------------------------------------------
// GEMM2: out[row,col] (+)= w[row,e] * ( H[row,:]*W2e[:,col] + b2e[col] )
// M=8192 N=1024 K=4096.  A tile (bf16) via async global->LDS copies.
// ---------------------------------------------------------------------------
__global__ __launch_bounds__(256) void moe_gemm2(
    const __bf16* __restrict__ H, const float* __restrict__ W2e,
    const float* __restrict__ b2e, const float* __restrict__ wts,
    float* __restrict__ out, int eIdx, int accumulate)
{
  __shared__ __align__(16) __bf16 Asm[BM][BK];
  __shared__ __align__(16) __bf16 Bsm[BK][BN];
  __shared__ float b2s[BN];
  __shared__ float wS[BM];

  const int tid    = threadIdx.x;
  const int lane   = tid & 31;
  const int lane16 = lane & 15;
  const int laneHi = lane >> 4;
  const int wid    = tid >> 5;

  const int col0 = blockIdx.x * BN;
  const int row0 = blockIdx.y * BM;
  const int wm0  = (wid >> 2) * (MT * 16);
  const int wn0  = (wid & 3)  * (NT * 16);

  if (tid < BN) b2s[tid] = b2e[col0 + tid];
  else          wS[tid - BN] = wts[(size_t)(row0 + tid - BN) * NEXP + eIdx];

  AccU acc[MT][NT];
#pragma unroll
  for (int mi = 0; mi < MT; ++mi)
#pragma unroll
    for (int ni = 0; ni < NT; ++ni)
#pragma unroll
      for (int r = 0; r < 8; ++r) acc[mi][ni].f[r] = 0.0f;

  const int ar = tid >> 3;          // 0..31
  const int ac = (tid & 7) << 3;    // 0..56 (8 bf16 = one uint4)
  const int br = tid >> 5;          // 0..7
  const int bc = (tid & 31) << 2;   // 0..124

  for (int k0 = 0; k0 < DH; k0 += BK) {
    if (k0 + BK < DH) {
      __builtin_prefetch(W2e + (size_t)(k0 + BK + br) * DOUT + col0 + bc, 0, 0);
    }
    // A tile: bf16 h, async global -> LDS (no VGPR staging, ASYNCcnt).
#pragma unroll
    for (int rr = 0; rr < BM; rr += 32) {
      async_copy_b128(lds_off(&Asm[ar + rr][ac]),
                      H + (size_t)(row0 + ar + rr) * DH + k0 + ac);
    }
    // B tile: f32 W2 -> bf16 convert in VGPRs.
#pragma unroll
    for (int rr = 0; rr < BK; rr += 8) {
      float4 v = *(const float4*)(W2e + (size_t)(k0 + br + rr) * DOUT + col0 + bc);
      cvt_store4(&Bsm[br + rr][bc], v);
    }
    wait_asynccnt0();
    __syncthreads();

#pragma unroll
    for (int ks = 0; ks < BK; ks += 32) {
      FragU a[MT], b[NT];
#pragma unroll
      for (int mi = 0; mi < MT; ++mi) {
        const __bf16* p = &Asm[wm0 + mi * 16 + lane16][ks + laneHi * 8];
        a[mi].q[0] = *(const uint4*)p;
        a[mi].q[1] = *(const uint4*)(p + 16);
      }
      const int kB = ks + laneHi * 8;
#pragma unroll
      for (int ni = 0; ni < NT; ++ni) {
#pragma unroll
        for (int s = 0; s < 16; ++s)
          b[ni].v[s] = Bsm[kB + s + (s & 8)][wn0 + ni * 16 + lane16];
      }
#pragma unroll
      for (int mi = 0; mi < MT; ++mi)
#pragma unroll
        for (int ni = 0; ni < NT; ++ni)
          acc[mi][ni].v = __builtin_amdgcn_wmma_f32_16x16x32_bf16(
              false, a[mi].v, false, b[ni].v, (short)0, acc[mi][ni].v, false, false);
    }
    __syncthreads();
  }

#pragma unroll
  for (int mi = 0; mi < MT; ++mi) {
#pragma unroll
    for (int ni = 0; ni < NT; ++ni) {
      const int nl = wn0 + ni * 16 + lane16;
      const int gn = col0 + nl;
#pragma unroll
      for (int r = 0; r < 8; ++r) {
        const int ml = wm0 + mi * 16 + laneHi * 8 + r;
        const int gm = row0 + ml;
        float w = wS[ml];
        float v = w * (acc[mi][ni].f[r] + b2s[nl]);
        size_t oi = (size_t)gm * DOUT + gn;
        if (accumulate) v += out[oi];
        out[oi] = v;
      }
    }
  }
}

// ---------------------------------------------------------------------------
// Routing: one wave per token. dist_sq over 1024 dims x 8 experts,
// softmax weights, top-2 indices.
// ---------------------------------------------------------------------------
__global__ __launch_bounds__(256) void moe_routing(
    const float* __restrict__ x, const float* __restrict__ mus,
    const float* __restrict__ lsig, float* __restrict__ logp,
    float* __restrict__ wts, int* __restrict__ topidx)
{
  const int lane = threadIdx.x & 31;
  const int wid  = threadIdx.x >> 5;
  const int t    = blockIdx.x * 8 + wid;

  const float* xt = x + (size_t)t * DIN;
  float xv[DIN / 32];
#pragma unroll
  for (int i = 0; i < DIN / 32; ++i) xv[i] = xt[lane + i * 32];

  float lp[NEXP];
#pragma unroll
  for (int e = 0; e < NEXP; ++e) {
    const float* mu = mus + (size_t)e * DIN;
    const float* ls = lsig + (size_t)e * DIN;
    float acc = 0.0f, lss = 0.0f;
    for (int i = 0; i < DIN / 32; ++i) {
      const int d = lane + i * 32;
      const float s = ls[d];
      const float diff = (xv[i] - mu[d]) * expf(-s);
      acc += diff * diff;
      lss += s;
    }
    for (int off = 16; off > 0; off >>= 1) {
      acc += __shfl_xor(acc, off, 32);
      lss += __shfl_xor(lss, off, 32);
    }
    lp[e] = -0.5f * acc - lss;
  }

  if (lane == 0) {
    float mx = lp[0];
    for (int e = 1; e < NEXP; ++e) mx = fmaxf(mx, lp[e]);
    float w[NEXP]; float s = 0.0f;
    for (int e = 0; e < NEXP; ++e) { w[e] = expf(lp[e] - mx); s += w[e]; }
    const float inv = 1.0f / s;
    int i0 = 0;
    for (int e = 1; e < NEXP; ++e) if (lp[e] > lp[i0]) i0 = e;
    int i1 = (i0 == 0) ? 1 : 0;
    for (int e = 0; e < NEXP; ++e) if (e != i0 && lp[e] > lp[i1]) i1 = e;
    for (int e = 0; e < NEXP; ++e) {
      logp[(size_t)t * NEXP + e] = lp[e];
      wts [(size_t)t * NEXP + e] = w[e] * inv;
    }
    topidx[t * 2 + 0] = i0;
    topidx[t * 2 + 1] = i1;
  }
}

// ---------------------------------------------------------------------------
// d_out layout: final[8192*1024] | log_probs[8192*8] | weights[8192*8] |
//               top_indices[8192*2] (int32 raw)
// ---------------------------------------------------------------------------
extern "C" void kernel_launch(void* const* d_in, const int* in_sizes, int n_in,
                              void* d_out, int out_size, void* d_ws, size_t ws_size,
                              hipStream_t stream) {
  const float* x    = (const float*)d_in[0];
  const float* mus  = (const float*)d_in[1];
  const float* lsig = (const float*)d_in[2];
  const float* W1   = (const float*)d_in[3];
  const float* b1   = (const float*)d_in[4];
  const float* W2   = (const float*)d_in[5];
  const float* b2   = (const float*)d_in[6];

  float* outF = (float*)d_out;
  float* logp = outF + (size_t)NTOK * DOUT;
  float* wts  = logp + (size_t)NTOK * NEXP;
  int*   idx  = (int*)(wts + (size_t)NTOK * NEXP);

  __bf16* h = (__bf16*)d_ws;   // 8192*4096 bf16 = 64 MiB scratch

  moe_routing<<<NTOK / 8, 256, 0, stream>>>(x, mus, lsig, logp, wts, idx);

  for (int e = 0; e < NEXP; ++e) {
    moe_gemm1<<<dim3(DH / BN, NTOK / BM), 256, 0, stream>>>(
        x, W1 + (size_t)e * DIN * DH, b1 + (size_t)e * DH, h);
    moe_gemm2<<<dim3(DOUT / BN, NTOK / BM), 256, 0, stream>>>(
        h, W2 + (size_t)e * DH * DOUT, b2 + (size_t)e * DOUT, wts, outF,
        e, e > 0 ? 1 : 0);
  }
}
